// HLGTransformer_61332132987001
// MI455X (gfx1250) — matrix operand order repaired and checked by reading, not compiled, not run on hardware
//
#include <hip/hip_runtime.h>
#include <hip/hip_bf16.h>

typedef __attribute__((ext_vector_type(16))) _Float16 v16h;
typedef __attribute__((ext_vector_type(8)))  _Float16 v8h;
typedef __attribute__((ext_vector_type(8)))  float    v8f;

#define XS 520   // x / y LDS row stride in halfs (512 + 8 pad, 16B-aligned rows)
#define QS 40    // q/k LDS row stride in halfs
#define VTS 72   // v-transposed LDS row stride in halfs
#define SS 68    // S LDS row stride in floats
#define PS 72    // P LDS row stride in halfs
#define SCALE 0.17677669529663687f  // 1/sqrt(32)

// ---------------- WMMA helpers (CDNA5 wave32, 16x16x32 f16 -> f32) ----------

__device__ __forceinline__ v8f wmma_f16(v16h a, v16h b, v8f c) {
  // (neg_a, A, neg_b, B, c_mod, C, reuse_a, reuse_b)
  return __builtin_amdgcn_wmma_f32_16x16x32_f16(false, a, false, b, (short)0, c,
                                                false, false);
}

// A-operand: 16x32 f16 tile from row-major LDS [row][k], stride in halfs.
// Lane l: row = row0 + l%16; 16 contiguous halfs at k0 + (l<16?0:16).
__device__ __forceinline__ v16h load_A(const _Float16* p0, int stride, int row0,
                                       int k0, int lane) {
  const _Float16* p = p0 + (row0 + (lane & 15)) * stride + k0 + ((lane >> 4) << 4);
  v8h lo = *(const v8h*)(p);
  v8h hi = *(const v8h*)(p + 8);
  return __builtin_shufflevector(lo, hi, 0,1,2,3,4,5,6,7,8,9,10,11,12,13,14,15);
}

// B-operand from pre-packed global weights: chunk (ntile, kblock) holds
// 32 lanes x 16 contiguous halfs (one 32B load per lane).
__device__ __forceinline__ v16h load_Bpk(const _Float16* __restrict__ w, int KB,
                                         int ntg, int kk, int lane) {
  return *(const v16h*)(w + ((((size_t)ntg * KB + kk) * 32 + lane) << 4));
}

// B-operand from row-major LDS where "rows" are the B columns (n) and halfs
// along the row are the K dim: lane l reads 16 contiguous halfs,
// n = n0 + l%16, k = k0 + (l/16)*16 + e.
__device__ __forceinline__ v16h load_Brows(const _Float16* p0, int stride, int n0,
                                           int k0, int lane) {
  return *(const v16h*)(p0 + (n0 + (lane & 15)) * stride + k0 + ((lane >> 4) << 4));
}

// ---------------- prep kernel: pack f32 weight [K][N] into WMMA-B f16 -------

__global__ __launch_bounds__(256) void pack_w_kernel(const float* __restrict__ src,
                                                     _Float16* __restrict__ dst,
                                                     int K, int N) {
  int KB = K >> 5;
  int total = K * N;
  for (int idx = blockIdx.x * blockDim.x + threadIdx.x; idx < total;
       idx += gridDim.x * blockDim.x) {
    int e  = idx & 15;
    int l  = (idx >> 4) & 31;
    int c  = idx >> 9;           // chunk = nt*KB + kb
    int kb = c % KB;
    int nt = c / KB;
    int k  = kb * 32 + ((l >> 4) << 4) + e;
    int n  = nt * 16 + (l & 15);
    dst[idx] = (_Float16)src[(size_t)k * N + n];
  }
}

// ---------------- prep kernel: DynamicPosBias MLP + gather ------------------

__global__ __launch_bounds__(256) void pos_bias_kernel(
    const float* __restrict__ pos_w, const float* __restrict__ pos_b,
    const float* __restrict__ g1, const float* __restrict__ b1,
    const float* __restrict__ w1, const float* __restrict__ c1,
    const float* __restrict__ g2, const float* __restrict__ b2,
    const float* __restrict__ w2, const float* __restrict__ c2,
    const float* __restrict__ g3, const float* __restrict__ b3,
    const float* __restrict__ w3, const float* __restrict__ c3,
    float* __restrict__ rpb)  // [16][64][64] padded
{
  __shared__ float sw1[1024], sw2[1024], sw3[512];
  __shared__ float sg1[32], sb1[32], sc1[32];
  __shared__ float sg2[32], sb2[32], sc2[32];
  __shared__ float sg3[32], sb3[32], sc3[16];
  __shared__ float spw[64], spb[32];
  __shared__ float spf[169 * 16];

  int tid = threadIdx.x;
  for (int i = tid; i < 1024; i += 256) { sw1[i] = w1[i]; sw2[i] = w2[i]; }
  for (int i = tid; i < 512;  i += 256) sw3[i] = w3[i];
  if (tid < 64) spw[tid] = pos_w[tid];
  if (tid < 32) {
    spb[tid] = pos_b[tid];
    sg1[tid] = g1[tid]; sb1[tid] = b1[tid]; sc1[tid] = c1[tid];
    sg2[tid] = g2[tid]; sb2[tid] = b2[tid]; sc2[tid] = c2[tid];
    sg3[tid] = g3[tid]; sb3[tid] = b3[tid];
  }
  if (tid < 16) sc3[tid] = c3[tid];
  __syncthreads();

  if (tid < 169) {
    float dy = (float)(tid / 13 - 6);
    float dx = (float)(tid % 13 - 6);
    float p[32], q[32];
    for (int j = 0; j < 32; ++j) p[j] = dy * spw[j] + dx * spw[32 + j] + spb[j];

    auto lnrelu = [](float* v, const float* g, const float* bb) {
      float m = 0.f;
      for (int j = 0; j < 32; ++j) m += v[j];
      m *= (1.f / 32.f);
      float var = 0.f;
      for (int j = 0; j < 32; ++j) { float d = v[j] - m; var += d * d; }
      var *= (1.f / 32.f);
      float rs = rsqrtf(var + 1e-5f);
      for (int j = 0; j < 32; ++j) {
        float t = (v[j] - m) * rs * g[j] + bb[j];
        v[j] = t > 0.f ? t : 0.f;
      }
    };

    lnrelu(p, sg1, sb1);
    for (int o = 0; o < 32; ++o) {
      float a = sc1[o];
      for (int j = 0; j < 32; ++j) a += p[j] * sw1[j * 32 + o];
      q[o] = a;
    }
    lnrelu(q, sg2, sb2);
    for (int o = 0; o < 32; ++o) {
      float a = sc2[o];
      for (int j = 0; j < 32; ++j) a += q[j] * sw2[j * 32 + o];
      p[o] = a;
    }
    lnrelu(p, sg3, sb3);
    for (int o = 0; o < 16; ++o) {
      float a = sc3[o];
      for (int j = 0; j < 32; ++j) a += p[j] * sw3[j * 16 + o];
      spf[tid * 16 + o] = a;
    }
  }
  __syncthreads();

  for (int idx = tid; idx < 16 * 64 * 64; idx += 256) {
    int h = idx >> 12;
    int rem = idx & 4095;
    int i = rem >> 6, j = rem & 63;
    float v = 0.f;
    if (i < 49 && j < 49) {
      int t = ((i / 7 - j / 7) + 6) * 13 + ((i % 7 - j % 7) + 6);
      v = spf[t * 16 + h];
    }
    rpb[idx] = v;
  }
}

// ---------------- fused per-window attention kernel -------------------------
// One workgroup (8 wave32) per batch window. All GEMMs via v_wmma_f32_16x16x32_f16.

__global__ __launch_bounds__(256) void attn_kernel(
    const float* __restrict__ x, const float* __restrict__ proj_b,
    const _Float16* __restrict__ qw16, const _Float16* __restrict__ kvw16,
    const _Float16* __restrict__ pw16, const float* __restrict__ rpb,
    float* __restrict__ out) {
  extern __shared__ char smem[];
  _Float16* sh_x  = (_Float16*)(smem);                 // [64][520] f16  66560 B
  _Float16* sh_y  = (_Float16*)(smem + 66560);         // [64][520] f16  66560 B
  _Float16* sh_q  = (_Float16*)(smem + 133120);        // [64][40]  f16   5120 B
  _Float16* sh_k  = (_Float16*)(smem + 138240);        // [64][40]  f16   5120 B
  _Float16* sh_vt = (_Float16*)(smem + 143360);        // [32][72]  f16   4608 B
  float*    sh_S  = (float*)   (smem + 147968);        // [64][68]  f32  17408 B
  _Float16* sh_P  = (_Float16*)(smem + 165376);        // [64][72]  f16   9216 B

  const int b    = blockIdx.x;
  const int tid  = threadIdx.x;
  const int lane = tid & 31;
  const int wv   = tid >> 5;

  // ---- phase 0: stage x[b] (49x512 f32) into LDS as f16, zero-pad to 64 rows
  const float* xb = x + (size_t)b * 49 * 512;
  for (int i = tid; i < 49 * 128; i += 256) {  // float4 granules
    float4 f = ((const float4*)xb)[i];
    int n = i >> 7;
    int c = (i & 127) << 2;
    _Float16* d = sh_x + n * XS + c;
    d[0] = (_Float16)f.x; d[1] = (_Float16)f.y;
    d[2] = (_Float16)f.z; d[3] = (_Float16)f.w;
  }
  for (int i = tid; i < 15 * 520; i += 256)
    sh_x[(49 + i / 520) * XS + (i % 520)] = (_Float16)0.f;
  __syncthreads();

  // ---- per-head fused attention
  for (int h = 0; h < 16; ++h) {
    // q/k/v head projections: 8 waves cover 4(M) x 2(N) tiles; K = 512
    {
      int mt = wv >> 1, ntl = wv & 1;
      int row0 = mt * 16, col0 = ntl * 16;
      int ntg_qk = 2 * h + ntl;       // q_w / kv_w "k" columns h*32..
      int ntg_v  = 32 + 2 * h + ntl;  // kv_w "v" columns 512+h*32..
      v8f aq = {}, ak = {}, av = {};
      for (int kk = 0; kk < 16; ++kk) {
        v16h a  = load_A(sh_x, XS, row0, kk * 32, lane);
        v16h bq = load_Bpk(qw16, 16, ntg_qk, kk, lane);
        v16h bk = load_Bpk(kvw16, 16, ntg_qk, kk, lane);
        v16h bv = load_Bpk(kvw16, 16, ntg_v, kk, lane);
        aq = wmma_f16(a, bq, aq);
        ak = wmma_f16(a, bk, ak);
        av = wmma_f16(a, bv, av);
      }
      for (int r = 0; r < 8; ++r) {
        int rr = row0 + ((lane >> 4) << 3) + r;
        int cc = col0 + (lane & 15);
        sh_q[rr * QS + cc] = (_Float16)aq[r];
        sh_k[rr * QS + cc] = (_Float16)ak[r];
        sh_vt[cc * VTS + rr] = (_Float16)av[r];  // store v transposed: [d][m]
      }
    }
    __syncthreads();

    // S = q k^T * scale + rpb[h] : 4x4 tiles, 2 per wave, K = 32 (one step)
    {
      const float* rp = rpb + ((size_t)h << 12);
      for (int t = wv; t < 16; t += 8) {
        int nt = t >> 2, mt = t & 3;
        v8f s = {};
        v16h a = load_A(sh_q, QS, nt * 16, 0, lane);
        v16h bb = load_Brows(sh_k, QS, mt * 16, 0, lane);  // B[d][m]=k[m][d]
        s = wmma_f16(a, bb, s);
        for (int r = 0; r < 8; ++r) {
          int rr = nt * 16 + ((lane >> 4) << 3) + r;
          int cc = mt * 16 + (lane & 15);
          sh_S[rr * SS + cc] = s[r] * SCALE + rp[(rr << 6) + cc];
        }
      }
    }
    __syncthreads();

    // masked softmax over the 49 valid columns; emit P as f16 (cols>=49 -> 0)
    if (tid < 64) {
      float* srow = sh_S + tid * SS;
      float mx = -1e30f;
      for (int j = 0; j < 49; ++j) mx = fmaxf(mx, srow[j]);
      float sum = 0.f;
      for (int j = 0; j < 49; ++j) {
        float e = __expf(srow[j] - mx);
        srow[j] = e;
        sum += e;
      }
      float rs = 1.f / sum;
      _Float16* prow = sh_P + tid * PS;
      for (int j = 0; j < 49; ++j) prow[j] = (_Float16)(srow[j] * rs);
      for (int j = 49; j < 64; ++j) prow[j] = (_Float16)0.f;
    }
    __syncthreads();

    // out_h = P @ v : 4(M) x 2(N) tiles, one per wave, K = 64 (two steps)
    {
      int nt = wv >> 1, dt = wv & 1;
      v8f o = {};
      for (int kk = 0; kk < 2; ++kk) {
        v16h a  = load_A(sh_P, PS, nt * 16, kk * 32, lane);
        v16h bb = load_Brows(sh_vt, VTS, dt * 16, kk * 32, lane);  // B[m][d]=v[m][d]
        o = wmma_f16(a, bb, o);
      }
      for (int r = 0; r < 8; ++r) {
        int rr = nt * 16 + ((lane >> 4) << 3) + r;
        int cc = dt * 16 + (lane & 15);
        sh_y[rr * XS + h * 32 + cc] = (_Float16)o[r];
      }
    }
    __syncthreads();
  }

  // ---- phase 2: out = y @ proj_w + proj_b, streamed to HBM (rows < 49)
  float* ob = out + (size_t)b * 49 * 512;
  for (int t = wv; t < 128; t += 8) {  // 4(M) x 32(N) tiles
    int mt = t >> 5, nt = t & 31;
    v8f acc = {};
    for (int kk = 0; kk < 16; ++kk) {
      v16h a  = load_A(sh_y, XS, mt * 16, kk * 32, lane);
      v16h bb = load_Bpk(pw16, 16, nt, kk, lane);
      acc = wmma_f16(a, bb, acc);
    }
    for (int r = 0; r < 8; ++r) {
      int rr = mt * 16 + ((lane >> 4) << 3) + r;
      int cc = nt * 16 + (lane & 15);
      if (rr < 49) ob[rr * 512 + cc] = acc[r] + proj_b[cc];
    }
  }
}

// ---------------- host launcher ---------------------------------------------

extern "C" void kernel_launch(void* const* d_in, const int* in_sizes, int n_in,
                              void* d_out, int out_size, void* d_ws, size_t ws_size,
                              hipStream_t stream) {
  const float* x      = (const float*)d_in[0];
  const float* q_w    = (const float*)d_in[1];
  const float* kv_w   = (const float*)d_in[2];
  const float* proj_w = (const float*)d_in[3];
  const float* proj_b = (const float*)d_in[4];
  const float* pos_w  = (const float*)d_in[5];
  const float* pos_b  = (const float*)d_in[6];
  const float* g1 = (const float*)d_in[7];
  const float* b1 = (const float*)d_in[8];
  const float* w1 = (const float*)d_in[9];
  const float* c1 = (const float*)d_in[10];
  const float* g2 = (const float*)d_in[11];
  const float* b2 = (const float*)d_in[12];
  const float* w2 = (const float*)d_in[13];
  const float* c2 = (const float*)d_in[14];
  const float* g3 = (const float*)d_in[15];
  const float* b3 = (const float*)d_in[16];
  const float* w3 = (const float*)d_in[17];
  const float* c3 = (const float*)d_in[18];
  float* out = (float*)d_out;

  const int Bn = in_sizes[0] / (49 * 512);

  // workspace layout: packed f16 weights + padded bias table
  char* ws = (char*)d_ws;
  _Float16* qw16  = (_Float16*)(ws);                               // 512*512 halfs
  _Float16* kvw16 = (_Float16*)(ws + 524288);                      // 512*1024 halfs
  _Float16* pw16  = (_Float16*)(ws + 524288 + 1048576);            // 512*512 halfs
  float*    rpb   = (float*)   (ws + 524288 + 1048576 + 524288);   // 16*64*64 f32

  pack_w_kernel<<<512, 256, 0, stream>>>(q_w, qw16, 512, 512);
  pack_w_kernel<<<1024, 256, 0, stream>>>(kv_w, kvw16, 512, 1024);
  pack_w_kernel<<<512, 256, 0, stream>>>(proj_w, pw16, 512, 512);
  pos_bias_kernel<<<1, 256, 0, stream>>>(pos_w, pos_b, g1, b1, w1, c1,
                                         g2, b2, w2, c2, g3, b3, w3, c3, rpb);

  const size_t lds_bytes = 174592;  // sh_x..sh_P partitioned in-kernel
  attn_kernel<<<Bn, 256, lds_bytes, stream>>>(x, proj_b, qw16, kvw16, pw16,
                                              rpb, out);
}
